// Relative_Attention_5669356835570
// MI455X (gfx1250) — matrix operand order repaired and checked by reading, not compiled, not run on hardware
//
#include <hip/hip_runtime.h>
#include <hip/hip_bf16.h>

// ---------------- problem constants ----------------
#define B_    32
#define N_    577
#define DIM_  768
#define NH_   12
#define HD_   64
#define K3_   2304
#define M_    (B_ * N_)           // 18464 = 577 * 32
#define QTILES ((N_ + 15) / 16)   // 37
#define QW    4                   // waves per attention block

typedef __attribute__((ext_vector_type(16))) __bf16 v16bf;
typedef __attribute__((ext_vector_type(8)))  float  v8f;

union AB16 {
    v16bf v;
    uint4 q[2];
    __bf16 e[16];
};

// ---------------- conversion kernels ----------------
__global__ void conv_f32_bf16(const float* __restrict__ src, __bf16* __restrict__ dst, size_t n) {
    size_t i = (size_t)blockIdx.x * blockDim.x + threadIdx.x;
    if (i < n) dst[i] = (__bf16)src[i];
}

// src[rows][cols] (f32) -> dst[cols][rows] (bf16); columns < scale_cols scaled
__global__ void conv_transpose_bf16(const float* __restrict__ src, __bf16* __restrict__ dst,
                                    int rows, int cols, int scale_cols, float scale) {
    size_t i = (size_t)blockIdx.x * blockDim.x + threadIdx.x;
    size_t total = (size_t)rows * cols;
    if (i >= total) return;
    int r = (int)(i / cols);
    int c = (int)(i % cols);
    float v = src[i];
    if (c < scale_cols) v *= scale;
    dst[(size_t)c * rows + r] = (__bf16)v;
}

// ---------------- bf16 WMMA GEMM, 32x64 tile per wave ----------------
// C[M,Nc] = A[M,K] * Bt[Nc,K]^T
// mode 0: bf16 C   mode 1: f32 C + bias   mode 2: scatter into q/k/v [B*h][N][64]
__global__ __launch_bounds__(256)
void gemm_bf16_wmma(const __bf16* __restrict__ A,
                    const __bf16* __restrict__ Bt,
                    __bf16* __restrict__ Cb,
                    float*  __restrict__ Cf,
                    const float* __restrict__ bias,
                    __bf16* __restrict__ Qd,
                    __bf16* __restrict__ Kd,
                    __bf16* __restrict__ Vd,
                    int mode, int M, int K, int Nc) {
    int lane = threadIdx.x & 31;
    int wave = threadIdx.x >> 5;
    int mt = blockIdx.y * 8 + wave;     // 32-row tile index
    int nt = blockIdx.x;                // 64-col tile index
    if (mt * 32 >= M) return;           // wave-uniform exit

    int m  = lane & 15;
    int hf = lane >> 4;

    const __bf16* arow[2];
    #pragma unroll
    for (int mi = 0; mi < 2; ++mi)
        arow[mi] = A + (size_t)(mt * 32 + mi * 16 + m) * K;
    const __bf16* brow[4];
    #pragma unroll
    for (int ni = 0; ni < 4; ++ni)
        brow[ni] = Bt + (size_t)(nt * 64 + ni * 16 + m) * K;

    v8f acc[2][4] = {};
    for (int kb = 0; kb < K; kb += 32) {
        // prefetch next chunk (speculative; safe OOB per ISA 10.5)
        __builtin_prefetch(arow[0] + kb + 64, 0, 1);
        __builtin_prefetch(brow[0] + kb + 64, 0, 1);
        AB16 a[2], b[4];
        #pragma unroll
        for (int mi = 0; mi < 2; ++mi) {
            // A 16x32 bf16 layout: lanes 0-15 K={kb+hf*8..+7, kb+16+hf*8..+7}
            a[mi].q[0] = *(const uint4*)(arow[mi] + kb + hf * 8);
            a[mi].q[1] = *(const uint4*)(arow[mi] + kb + 16 + hf * 8);
        }
        #pragma unroll
        for (int ni = 0; ni < 4; ++ni) {
            // B 32x16 bf16 layout: lane col = m, K={kb+hf*8..+7, kb+16+hf*8..+7}
            b[ni].q[0] = *(const uint4*)(brow[ni] + kb + hf * 8);
            b[ni].q[1] = *(const uint4*)(brow[ni] + kb + 16 + hf * 8);
        }
        #pragma unroll
        for (int mi = 0; mi < 2; ++mi)
            #pragma unroll
            for (int ni = 0; ni < 4; ++ni)
                acc[mi][ni] = __builtin_amdgcn_wmma_f32_16x16x32_bf16(
                    false, a[mi].v, false, b[ni].v, (short)0, acc[mi][ni], false, false);
    }

    if (mode == 2) {
        // scatter into q/k/v [B*h][N][64]; col -> (which, h, d); 64-col tiles never straddle
        #pragma unroll
        for (int ni = 0; ni < 4; ++ni) {
            int nc = nt * 64 + ni * 16 + m;
            int which = nc / DIM_;
            int rem   = nc % DIM_;
            int h = rem / HD_, d = rem % HD_;
            __bf16* dstbuf = (which == 0) ? Qd : ((which == 1) ? Kd : Vd);
            #pragma unroll
            for (int mi = 0; mi < 2; ++mi) {
                #pragma unroll
                for (int r = 0; r < 8; ++r) {
                    int row = mt * 32 + mi * 16 + r + 8 * hf;
                    int bb = row / N_, n = row % N_;
                    dstbuf[((size_t)(bb * NH_ + h) * N_ + n) * HD_ + d] =
                        (__bf16)acc[mi][ni][r];
                }
            }
        }
    } else if (mode == 1) {
        #pragma unroll
        for (int ni = 0; ni < 4; ++ni) {
            int nc = nt * 64 + ni * 16 + m;
            float bv = bias ? bias[nc] : 0.0f;
            #pragma unroll
            for (int mi = 0; mi < 2; ++mi) {
                #pragma unroll
                for (int r = 0; r < 8; ++r) {
                    int row = mt * 32 + mi * 16 + r + 8 * hf;
                    Cf[(size_t)row * Nc + nc] = acc[mi][ni][r] + bv;
                }
            }
        }
    } else {
        #pragma unroll
        for (int ni = 0; ni < 4; ++ni) {
            int nc = nt * 64 + ni * 16 + m;
            #pragma unroll
            for (int mi = 0; mi < 2; ++mi) {
                #pragma unroll
                for (int r = 0; r < 8; ++r) {
                    int row = mt * 32 + mi * 16 + r + 8 * hf;
                    Cb[(size_t)row * Nc + nc] = (__bf16)acc[mi][ni][r];
                }
            }
        }
    }
}

// ---------------- fused flash attention with relative-position bias ----------------
__global__ __launch_bounds__(128)
void attn_wmma(const __bf16* __restrict__ Q,       // [BH][N][64] (pre-scaled)
               const __bf16* __restrict__ Kmat,    // [BH][N][64]
               const __bf16* __restrict__ Vmat,    // [BH][N][64]
               const float*  __restrict__ bias_table, // [TSZ][12]
               const int*    __restrict__ rel_index,  // [N][N]
               __bf16* __restrict__ Out) {            // [B*N][768]
    __shared__ __attribute__((aligned(16))) __bf16 ldsK[32][64];   // keys x d
    __shared__ __attribute__((aligned(16))) __bf16 ldsVt[64][32];  // d x keys
    __shared__ __attribute__((aligned(16))) __bf16 ldsP[QW][16][32];

    int tid  = threadIdx.x;
    int lane = tid & 31;
    int wave = tid >> 5;
    int bh = blockIdx.y;
    int b = bh / NH_, h = bh % NH_;
    int qtile = blockIdx.x * QW + wave;
    bool active = qtile < QTILES;

    int m  = lane & 15;
    int hf = lane >> 4;

    const __bf16* qbase = Q    + (size_t)bh * N_ * HD_;
    const __bf16* kbase = Kmat + (size_t)bh * N_ * HD_;
    const __bf16* vbase = Vmat + (size_t)bh * N_ * HD_;

    // preload this wave's Q rows into A layout (two 32-wide k-steps over d=64)
    AB16 aq[2];
    {
        int qr = qtile * 16 + m;
        if (!active || qr >= N_) qr = N_ - 1;
        const __bf16* qrow = qbase + (size_t)qr * HD_;
        #pragma unroll
        for (int ds = 0; ds < 2; ++ds) {
            aq[ds].q[0] = *(const uint4*)(qrow + ds * 32 + hf * 8);
            aq[ds].q[1] = *(const uint4*)(qrow + ds * 32 + 16 + hf * 8);
        }
    }

    float mrow[8], lrow[8];
    #pragma unroll
    for (int r = 0; r < 8; ++r) { mrow[r] = -1e30f; lrow[r] = 0.0f; }
    v8f o[4] = {};

    const int nkt = (N_ + 31) / 32;  // 19 key tiles
    for (int kt = 0; kt < nkt; ++kt) {
        __syncthreads();
        // ---- cooperative staging: K tile and V^T tile (128 threads) ----
        {
            int key  = tid >> 2;          // 0..31
            int dseg = (tid & 3) * 16;    // 0,16,32,48
            int kg = kt * 32 + key;
            int kc = (kg < N_) ? kg : (N_ - 1);
            AB16 tk, tv;
            tk.q[0] = *(const uint4*)(kbase + (size_t)kc * HD_ + dseg);
            tk.q[1] = *(const uint4*)(kbase + (size_t)kc * HD_ + dseg + 8);
            *(uint4*)&ldsK[key][dseg]     = tk.q[0];
            *(uint4*)&ldsK[key][dseg + 8] = tk.q[1];
            tv.q[0] = *(const uint4*)(vbase + (size_t)kc * HD_ + dseg);
            tv.q[1] = *(const uint4*)(vbase + (size_t)kc * HD_ + dseg + 8);
            #pragma unroll
            for (int j = 0; j < 16; ++j) ldsVt[dseg + j][key] = tv.e[j];
        }
        __syncthreads();

        if (active) {
            // ---- S = Q K^T over 32 keys: two 16-key C tiles, 2 k-steps each ----
            v8f s[2];
            #pragma unroll
            for (int k2 = 0; k2 < 2; ++k2) {
                v8f sa = {};
                int keyl = k2 * 16 + m;
                #pragma unroll
                for (int ds = 0; ds < 2; ++ds) {
                    AB16 bmat;
                    bmat.q[0] = *(const uint4*)(&ldsK[keyl][ds * 32 + hf * 8]);
                    bmat.q[1] = *(const uint4*)(&ldsK[keyl][ds * 32 + 16 + hf * 8]);
                    sa = __builtin_amdgcn_wmma_f32_16x16x32_bf16(false, aq[ds].v, false, bmat.v,
                                                                 (short)0, sa, false, false);
                }
                s[k2] = sa;
            }
            // ---- add relative-position bias, mask out-of-range ----
            #pragma unroll
            for (int k2 = 0; k2 < 2; ++k2) {
                int key = kt * 32 + k2 * 16 + m;
                #pragma unroll
                for (int r = 0; r < 8; ++r) {
                    int qi = qtile * 16 + r + 8 * hf;
                    if (key < N_ && qi < N_) {
                        int idx = rel_index[qi * N_ + key];
                        s[k2][r] += bias_table[idx * NH_ + h];
                    } else {
                        s[k2][r] = -1e30f;
                    }
                }
            }
            // ---- online softmax: row stats via 16-lane xor-shuffle reductions ----
            float p0[8], p1[8];
            #pragma unroll
            for (int r = 0; r < 8; ++r) {
                float tmax = fmaxf(s[0][r], s[1][r]);
                #pragma unroll
                for (int off = 1; off < 16; off <<= 1)
                    tmax = fmaxf(tmax, __shfl_xor(tmax, off, 32));
                float mn = fmaxf(mrow[r], tmax);
                float alpha = __expf(mrow[r] - mn);
                float e0 = __expf(s[0][r] - mn);
                float e1 = __expf(s[1][r] - mn);
                float rs = e0 + e1;
                #pragma unroll
                for (int off = 1; off < 16; off <<= 1)
                    rs += __shfl_xor(rs, off, 32);
                lrow[r] = lrow[r] * alpha + rs;
                mrow[r] = mn;
                p0[r] = e0; p1[r] = e1;
                #pragma unroll
                for (int dt = 0; dt < 4; ++dt) o[dt][r] *= alpha;
            }
            // ---- transpose P (C layout -> A layout) through this wave's LDS slab ----
            #pragma unroll
            for (int r = 0; r < 8; ++r) {
                ldsP[wave][r + 8 * hf][m]      = (__bf16)p0[r];
                ldsP[wave][r + 8 * hf][16 + m] = (__bf16)p1[r];
            }
            AB16 ap;
            ap.q[0] = *(const uint4*)(&ldsP[wave][m][hf * 8]);
            ap.q[1] = *(const uint4*)(&ldsP[wave][m][16 + hf * 8]);
            // ---- O += P @ V  (four 16-col d tiles) ----
            #pragma unroll
            for (int dt = 0; dt < 4; ++dt) {
                AB16 bmat;
                bmat.q[0] = *(const uint4*)(&ldsVt[dt * 16 + m][hf * 8]);
                bmat.q[1] = *(const uint4*)(&ldsVt[dt * 16 + m][16 + hf * 8]);
                o[dt] = __builtin_amdgcn_wmma_f32_16x16x32_bf16(false, ap.v, false, bmat.v,
                                                                (short)0, o[dt], false, false);
            }
        }
    }

    // ---- epilogue: normalize and store bf16 [b*N+qi][h*64 + d] ----
    if (active) {
        #pragma unroll
        for (int r = 0; r < 8; ++r) {
            int qi = qtile * 16 + r + 8 * hf;
            if (qi < N_) {
                float inv = (lrow[r] > 0.0f) ? (1.0f / lrow[r]) : 0.0f;
                #pragma unroll
                for (int dt = 0; dt < 4; ++dt) {
                    Out[(size_t)(b * N_ + qi) * DIM_ + h * HD_ + dt * 16 + m] =
                        (__bf16)(o[dt][r] * inv);
                }
            }
        }
    }
}

// ---------------- host-side launch ----------------
extern "C" void kernel_launch(void* const* d_in, const int* in_sizes, int n_in,
                              void* d_out, int out_size, void* d_ws, size_t ws_size,
                              hipStream_t stream) {
    const float* x          = (const float*)d_in[0];
    const float* qkv_w      = (const float*)d_in[1];
    const float* bias_table = (const float*)d_in[2];
    const float* proj_w     = (const float*)d_in[3];
    const float* proj_b     = (const float*)d_in[4];
    const int*   rel_index  = (const int*)d_in[5];
    float* out = (float*)d_out;

    char* ws = (char*)d_ws;
    size_t off = 0;
    auto alloc = [&](size_t bytes) -> void* {
        void* p = ws + off;
        off += (bytes + 255) & ~(size_t)255;
        return p;
    };
    __bf16* xb      = (__bf16*)alloc((size_t)M_ * DIM_ * 2);
    __bf16* qkvwT   = (__bf16*)alloc((size_t)K3_ * DIM_ * 2);
    __bf16* projwT  = (__bf16*)alloc((size_t)DIM_ * DIM_ * 2);
    __bf16* qb      = (__bf16*)alloc((size_t)B_ * NH_ * N_ * HD_ * 2);
    __bf16* kb      = (__bf16*)alloc((size_t)B_ * NH_ * N_ * HD_ * 2);
    __bf16* vb      = (__bf16*)alloc((size_t)B_ * NH_ * N_ * HD_ * 2);
    __bf16* attn_o  = (__bf16*)alloc((size_t)M_ * DIM_ * 2);

    // 1) bf16 conversions: x; transposed weights (q-scale 1/8 folded into qkv_w cols 0..767)
    {
        size_t n = (size_t)M_ * DIM_;
        conv_f32_bf16<<<dim3((unsigned)((n + 255) / 256)), 256, 0, stream>>>(x, xb, n);
        size_t nq = (size_t)DIM_ * K3_;
        conv_transpose_bf16<<<dim3((unsigned)((nq + 255) / 256)), 256, 0, stream>>>(
            qkv_w, qkvwT, DIM_, K3_, DIM_, 0.125f);
        size_t np = (size_t)DIM_ * DIM_;
        conv_transpose_bf16<<<dim3((unsigned)((np + 255) / 256)), 256, 0, stream>>>(
            proj_w, projwT, DIM_, DIM_, 0, 1.0f);
    }

    // 2) fused qkv GEMM + scatter: [18464,768] x [768,2304] -> q/k/v [B*h][N][64]
    {
        dim3 grid(K3_ / 64, (M_ / 32 + 7) / 8);   // (36, 73)
        gemm_bf16_wmma<<<grid, 256, 0, stream>>>(xb, qkvwT, nullptr, nullptr, nullptr,
                                                 qb, kb, vb, /*mode=*/2, M_, DIM_, K3_);
    }

    // 3) fused flash attention with relative-position bias
    {
        dim3 grid((QTILES + QW - 1) / QW, B_ * NH_);  // (10, 384)
        attn_wmma<<<grid, 128, 0, stream>>>(qb, kb, vb, bias_table, rel_index, attn_o);
    }

    // 4) output projection: [18464,768] x [768,768] + bias -> f32 d_out
    {
        dim3 grid(DIM_ / 64, (M_ / 32 + 7) / 8);  // (12, 73)
        gemm_bf16_wmma<<<grid, 256, 0, stream>>>(attn_o, projwT, nullptr, out, proj_b,
                                                 nullptr, nullptr, nullptr, /*mode=*/1,
                                                 M_, DIM_, DIM_);
    }
}
